// Mamba_61744449847385
// MI455X (gfx1250) — matrix operand order repaired and checked by reading, not compiled, not run on hardware
//
#include <hip/hip_runtime.h>
#include <hip/hip_bf16.h>

// ---------------- problem constants ----------------
#define DMODEL 1024
#define DI     2048   // d_inner
#define NST    16     // d_state
#define DCONV  4
#define DTR    64     // dt_rank
#define XDBL   96     // dt_rank + 2*d_state
#define BATCH  2
#define SEQL   2048
#define NTOK   (BATCH * SEQL)   // 4096 tokens

// ---------------- WMMA types (CDNA5 / gfx1250, wave32) ----------------
typedef __attribute__((ext_vector_type(16))) __bf16 v16bf;
typedef __attribute__((ext_vector_type(8)))  float  v8f;

union FragBF { v16bf v; unsigned u[8]; };

__device__ __forceinline__ unsigned pack_bf16(float a, float b) {
    // round-to-nearest-even f32 -> bf16, pack (lo=a, hi=b)
    unsigned ua = __float_as_uint(a); ua += 0x7FFFu + ((ua >> 16) & 1u);
    unsigned ub = __float_as_uint(b); ub += 0x7FFFu + ((ub >> 16) & 1u);
    return (ua >> 16) | (ub & 0xFFFF0000u);
}

// A-fragment 16x32 bf16: lane = M%16 (+half); dword v holds K-pair
//   v<4:  K = 2v + 8*half ;  v>=4: K = 16 + 2(v-4) + 8*half
__device__ __forceinline__ v16bf load_a_frag(const unsigned* __restrict__ sA,
                                             int m_off, int lane) {
    FragBF f;
    const unsigned* rowp = sA + (m_off + (lane & 15)) * 17;
    const int h4 = (lane >> 4) << 2;
#pragma unroll
    for (int v = 0; v < 8; ++v) {
        int pair = v + h4 + ((v >= 4) ? 4 : 0);
        f.u[v] = rowp[pair];
    }
    return f.v;
}

// B-fragment 32x16 bf16: lane = N%16; dword v holds K-pair  (v<4: K = 2v + 8*half; v>=4: K = 16 + 2(v-4) + 8*half)
__device__ __forceinline__ v16bf load_b_frag(const unsigned* __restrict__ sB,
                                             int n_off, int lane) {
    FragBF f;
    const unsigned* rowp = sB + (n_off + (lane & 15)) * 17;
    const int h4 = (lane >> 4) << 2;
#pragma unroll
    for (int v = 0; v < 8; ++v) f.u[v] = rowp[v + h4 + ((v >= 4) ? 4 : 0)];
    return f.v;
}

// ---------------- GEMM:  C[M,N] = A[M,K] * W[N,K]^T  (+ epilogue) ----------------
// BM=128, BN=128, BK=32; 256 threads = 8 waves as 4x2 wave-tiles of 32x64,
// each wave holds 2x4 v_wmma_f32_16x16x32_bf16 accumulators (8 WMMA / K-step).
#define BM 128
#define BN 128
#define BK 32

__global__ __launch_bounds__(256)
void gemm_bf16w(const float* __restrict__ A, int lda,
                const float* __restrict__ W, int ldw,
                float* __restrict__ C, int ldc,
                int M, int N, int K,
                const float* __restrict__ bias, int mode) {
    __shared__ unsigned sA[BM * 17];   // bf16 pairs, padded stride (bank-conflict free)
    __shared__ unsigned sB[BN * 17];

    const int tid  = threadIdx.x;
    const int m0   = blockIdx.y * BM;
    const int n0   = blockIdx.x * BN;
    const int lane = tid & 31;
    const int wave = tid >> 5;
    const int wm   = wave & 3;         // 0..3 row wave-tile (32 rows)
    const int wn   = wave >> 2;        // 0..1 col wave-tile (64 cols)

    // staging coordinates: 8 pairs per thread for each of A and B tiles
    const int srow = tid >> 4;         // 0..15 base row (stride 16 over i)
    const int scp  = tid & 15;         // pair column 0..15

    const bool fullA = (m0 + BM <= M); // K is checked per k0 (all calls: multiple of BK)
    const bool fullB = (n0 + BN <= N);

    v8f acc[2][4] = {};

    for (int k0 = 0; k0 < K; k0 += BK) {
        const bool fullK = (k0 + BK <= K);
        // ---- stage A tile (BM x BK) as packed bf16 pairs ----
        if (fullA && fullK) {
#pragma unroll
            for (int i = 0; i < 8; ++i) {
                int row = srow + i * 16;
                const float2 f = *(const float2*)&A[(size_t)(m0 + row) * lda + k0 + scp * 2];
                sA[row * 17 + scp] = pack_bf16(f.x, f.y);
            }
        } else {
#pragma unroll
            for (int i = 0; i < 8; ++i) {
                int row = srow + i * 16;
                int gm = m0 + row, gk = k0 + scp * 2;
                float f0 = 0.f, f1 = 0.f;
                if (gm < M && gk < K) {
                    f0 = A[(size_t)gm * lda + gk];
                    if (gk + 1 < K) f1 = A[(size_t)gm * lda + gk + 1];
                }
                sA[row * 17 + scp] = pack_bf16(f0, f1);
            }
        }
        // ---- stage W tile (BN x BK) ----
        if (fullB && fullK) {
#pragma unroll
            for (int i = 0; i < 8; ++i) {
                int row = srow + i * 16;
                const float2 f = *(const float2*)&W[(size_t)(n0 + row) * ldw + k0 + scp * 2];
                sB[row * 17 + scp] = pack_bf16(f.x, f.y);
            }
        } else {
#pragma unroll
            for (int i = 0; i < 8; ++i) {
                int row = srow + i * 16;
                int gn = n0 + row, gk = k0 + scp * 2;
                float f0 = 0.f, f1 = 0.f;
                if (gn < N && gk < K) {
                    f0 = W[(size_t)gn * ldw + gk];
                    if (gk + 1 < K) f1 = W[(size_t)gn * ldw + gk + 1];
                }
                sB[row * 17 + scp] = pack_bf16(f0, f1);
            }
        }
        // speculative prefetch of next A k-slab (global_prefetch_b8)
        if (k0 + BK < K && m0 + (tid & 127) < M)
            __builtin_prefetch(&A[(size_t)(m0 + (tid & 127)) * lda + k0 + BK], 0, 0);
        __syncthreads();

        v16bf af[2], bfr[4];
        af[0]  = load_a_frag(sA, wm * 32,      lane);
        af[1]  = load_a_frag(sA, wm * 32 + 16, lane);
#pragma unroll
        for (int j = 0; j < 4; ++j)
            bfr[j] = load_b_frag(sB, wn * 64 + j * 16, lane);
#pragma unroll
        for (int i = 0; i < 2; ++i)
#pragma unroll
            for (int j = 0; j < 4; ++j)
                acc[i][j] = __builtin_amdgcn_wmma_f32_16x16x32_bf16(
                    false, af[i], false, bfr[j], (short)0, acc[i][j], false, false);
        __syncthreads();
    }

    // ---- epilogue: C/D layout VGPR r -> M = r + 8*half, N = lane%16 ----
    const int half = lane >> 4, nn = lane & 15;
    const bool fullC = fullA && fullB;
#pragma unroll
    for (int i = 0; i < 2; ++i)
#pragma unroll
        for (int j = 0; j < 4; ++j) {
            int ncol = n0 + wn * 64 + j * 16 + nn;
#pragma unroll
            for (int r = 0; r < 8; ++r) {
                int mrow = m0 + wm * 32 + i * 16 + r + 8 * half;
                if (fullC || (mrow < M && ncol < N)) {
                    float v = acc[i][j][r];
                    if (mode == 1) {           // softplus(v + bias[n])
                        v += bias[ncol];
                        v = (v > 20.f) ? v : log1pf(__expf(v));
                    }
                    C[(size_t)mrow * ldc + ncol] = v;
                }
            }
        }
}

// ---------------- causal depthwise conv1d (k=4) + bias + SiLU ----------------
__global__ __launch_bounds__(256)
void conv_silu(const float* __restrict__ xz, const float* __restrict__ cw,
               const float* __restrict__ cb, float* __restrict__ xact) {
    int idx = blockIdx.x * 256 + threadIdx.x;          // over B*L*DI
    if (idx >= NTOK * DI) return;
    int d  = idx & (DI - 1);
    int bl = idx >> 11;                                 // / DI
    int l  = bl & (SEQL - 1);
    int b  = bl >> 11;                                  // / SEQL
    float acc = cb[d];
#pragma unroll
    for (int j = 0; j < DCONV; ++j) {
        int t = l - (DCONV - 1) + j;
        if (t >= 0)
            acc += cw[d * DCONV + j] * xz[((size_t)(b * SEQL + t) * (2 * DI)) + d];
    }
    xact[idx] = acc / (1.f + __expf(-acc));             // SiLU
}

// ---------------- selective scan (sequential in t, parallel over B x D) ----------------
// one thread per channel d; 16 fp32 states in registers; B/C staged via LDS
__global__ __launch_bounds__(256)
void scan_kernel(const float* __restrict__ xz,   const float* __restrict__ xact,
                 const float* __restrict__ xdbl, const float* __restrict__ dt,
                 const float* __restrict__ Alog, const float* __restrict__ Dp,
                 float* __restrict__ ybuf) {
    __shared__ float sB[32][NST];
    __shared__ float sC[32][NST];
    const int tid = threadIdx.x;
    const int d   = blockIdx.x * 256 + tid;
    const int b   = blockIdx.y;

    float a[NST], h[NST];
#pragma unroll
    for (int n = 0; n < NST; ++n) {
        a[n] = -__expf(Alog[d * NST + n]);
        h[n] = 0.f;
    }
    const float Dd = Dp[d];

    for (int t0 = 0; t0 < SEQL; t0 += 32) {
        __syncthreads();
#pragma unroll
        for (int i = 0; i < 4; ++i) {                   // 32 t x 32 (B|C) = 1024 floats
            int e  = tid + i * 256;
            int tt = e >> 5, j = e & 31;
            float v = xdbl[(size_t)(b * SEQL + t0 + tt) * XDBL + DTR + j];
            if (j < NST) sB[tt][j] = v; else sC[tt][j - NST] = v;
        }
        __syncthreads();
        for (int tt = 0; tt < 32; ++tt) {
            const size_t base = (size_t)(b * SEQL + t0 + tt);
            const float dtv = dt[base * DI + d];
            const float xv  = xact[base * DI + d];
            const float ubx = dtv * xv;
            float y = 0.f;
#pragma unroll
            for (int n = 0; n < NST; ++n) {
                h[n] = __expf(dtv * a[n]) * h[n] + ubx * sB[tt][n];
                y   += h[n] * sC[tt][n];
            }
            const float zv = xz[base * (2 * DI) + DI + d];
            ybuf[base * DI + d] = (y + xv * Dd) * (zv / (1.f + __expf(-zv)));
        }
    }
}

// ---------------- driver ----------------
extern "C" void kernel_launch(void* const* d_in, const int* in_sizes, int n_in,
                              void* d_out, int out_size, void* d_ws, size_t ws_size,
                              hipStream_t stream) {
    const float* hidden = (const float*)d_in[0];
    const float* Win    = (const float*)d_in[1];
    const float* convW  = (const float*)d_in[2];
    const float* convB  = (const float*)d_in[3];
    const float* Wx     = (const float*)d_in[4];
    const float* Wdt    = (const float*)d_in[5];
    const float* bdt    = (const float*)d_in[6];
    const float* Alog   = (const float*)d_in[7];
    const float* Dp     = (const float*)d_in[8];
    const float* Wout   = (const float*)d_in[9];
    float* out = (float*)d_out;

    float* w    = (float*)d_ws;
    float* xz   = w;                                   // NTOK x 2*DI
    float* xact = xz   + (size_t)NTOK * 2 * DI;        // NTOK x DI  (y written in place)
    float* xdbl = xact + (size_t)NTOK * DI;            // NTOK x 96
    float* dtb  = xdbl + (size_t)NTOK * XDBL;          // NTOK x DI

    dim3 blk(256);

    // 1) xz = hidden @ Win^T         M=4096 N=4096 K=1024
    gemm_bf16w<<<dim3((2 * DI) / BN, NTOK / BM), blk, 0, stream>>>(
        hidden, DMODEL, Win, DMODEL, xz, 2 * DI, NTOK, 2 * DI, DMODEL, nullptr, 0);

    // 2) causal conv + SiLU
    conv_silu<<<dim3((NTOK * DI) / 256), blk, 0, stream>>>(xz, convW, convB, xact);

    // 3) x_dbl = x @ Wx^T            M=4096 N=96 K=2048
    gemm_bf16w<<<dim3((XDBL + BN - 1) / BN, NTOK / BM), blk, 0, stream>>>(
        xact, DI, Wx, DI, xdbl, XDBL, NTOK, XDBL, DI, nullptr, 0);

    // 4) dt = softplus(x_dbl[:, :64] @ Wdt^T + bdt)   M=4096 N=2048 K=64 (lda=96)
    gemm_bf16w<<<dim3(DI / BN, NTOK / BM), blk, 0, stream>>>(
        xdbl, XDBL, Wdt, DTR, dtb, DI, NTOK, DI, DTR, bdt, 1);

    // 5) selective scan (+ x*D, * silu(z)); y written over xact
    scan_kernel<<<dim3(DI / 256, BATCH), blk, 0, stream>>>(
        xz, xact, xdbl, dtb, Alog, Dp, xact);

    // 6) out = y @ Wout^T            M=4096 N=1024 K=2048
    gemm_bf16w<<<dim3(DMODEL / BN, NTOK / BM), blk, 0, stream>>>(
        xact, DI, Wout, DI, out, DMODEL, NTOK, DMODEL, DI, nullptr, 0);
}
